// AttentionWithEncoderKV_48593259987179
// MI455X (gfx1250) — matrix operand order repaired and checked by reading, not compiled, not run on hardware
//
#include <hip/hip_runtime.h>
#include <hip/hip_bf16.h>

typedef __attribute__((ext_vector_type(16))) __bf16 v16bf;
typedef __attribute__((ext_vector_type(8)))  __bf16 v8bf;
typedef __attribute__((ext_vector_type(8)))  float  v8f;

#define WMMA_BF16(a, b, c) \
  __builtin_amdgcn_wmma_f32_16x16x32_bf16(false, (a), false, (b), (short)0, (c), false, false)

// Problem constants
constexpr int BATCH = 4;
constexpr int HEADS = 16;
constexpr int NSEQ  = 1024;
constexpr int DMODEL= 1152;     // H * hd
constexpr int HD    = 72;       // head dim
constexpr int HDP   = 96;       // padded head dim (3 chunks of 32)
constexpr int BH    = BATCH * HEADS;

// ---------------------------------------------------------------------------
// CDNA5 async global->LDS copy (16B per op) + async counter waits.
// LDS byte address = low 32 bits of the generic shared pointer (ISA 10.2:
// LDS aperture maps addr[31:0] to the wave's LDS allocation).
__device__ inline void async_copy16(unsigned lds_off, const __bf16* g) {
  unsigned long long ga = (unsigned long long)(const void*)g;
  asm volatile("global_load_async_to_lds_b128 %0, %1, off"
               :: "v"(lds_off), "v"(ga) : "memory");
}
#define S_WAIT_ASYNC_6() asm volatile("s_wait_asynccnt 6" ::: "memory")
#define S_WAIT_ASYNC_0() asm volatile("s_wait_asynccnt 0" ::: "memory")

// ---------------------------------------------------------------------------
// Fragment loaders (WMMA operands, wave32; this code's contiguous k order, same in A and B)
// A 16x32 bf16: lane L -> row r=L&15, half h=L>>4; elems i -> K = h*16 + i
// (contiguous).
__device__ inline v16bf load_a16x32(const __bf16* __restrict__ A, int lda, int lane) {
  int r = lane & 15, h = lane >> 4;
  const __bf16* p = A + (size_t)r * lda + h * 16;
  v8bf lo = *(const v8bf*)(p);
  v8bf hi = *(const v8bf*)(p + 8);
  v16bf out;
#pragma unroll
  for (int i = 0; i < 8; ++i) { out[i] = lo[i]; out[i + 8] = hi[i]; }
  return out;
}

// B 32x16 bf16 where B[k][n] = W[n0+n][k0+k] (W row-major, C = A @ W^T form):
// lane L -> col n=L&15, half h=L>>4; elems i -> K = h*16 + i (contiguous).
__device__ inline v16bf load_bT32x16(const __bf16* __restrict__ W, int ldw, int lane) {
  int n = lane & 15, h = lane >> 4;
  return *(const v16bf*)(W + (size_t)n * ldw + h * 16);
}

// ---------------------------------------------------------------------------
// fp32 -> bf16 elementwise convert
__global__ __launch_bounds__(256) void cvt_f32_bf16(const float* __restrict__ in,
                                                    __bf16* __restrict__ out, int n) {
  int i = blockIdx.x * 256 + threadIdx.x;
  if (i < n) out[i] = (__bf16)in[i];
}

// enc_k -> padded (BH, NSEQ, HDP) bf16 ; enc_v -> transposed (BH, HDP, NSEQ) bf16
__global__ __launch_bounds__(256) void enc_convert(const float* __restrict__ ek,
                                                   const float* __restrict__ ev,
                                                   __bf16* __restrict__ ekb,
                                                   __bf16* __restrict__ evT) {
  int i = blockIdx.x * 256 + threadIdx.x;
  const int total = BH * NSEQ * HD;
  if (i >= total) return;
  int d  = i % HD;
  int r  = i / HD;
  int ns = r % NSEQ;
  int bh = r / NSEQ;
  ekb[((size_t)bh * NSEQ + ns) * HDP + d] = (__bf16)ek[i];
  evT[((size_t)bh * HDP + d) * NSEQ + ns] = (__bf16)ev[i];
}

// ---------------------------------------------------------------------------
// QKV GEMM: qkv[g, c] = x[g, :] . qkv_w[c, :] + qkv_b[c]
// M = 4096 (B*N), N = 3456 (3*D), K = 1152.
// One wave: 32x64 tile (2 A-frags x 4 B-frags = 8 WMMA / 32-K step),
// register double-buffered so loads overlap WMMA.
__global__ __launch_bounds__(256) void qkv_gemm(const __bf16* __restrict__ xb,
                                                const __bf16* __restrict__ wb,
                                                const float*  __restrict__ bias,
                                                __bf16* __restrict__ qb,
                                                __bf16* __restrict__ kb,
                                                __bf16* __restrict__ vT) {
  const int NQ = (3 * DMODEL) / 64;                // 54
  int gid  = blockIdx.x * 8 + (threadIdx.x >> 5);  // wave id
  int lane = threadIdx.x & 31;
  int mt = gid / NQ;                               // 0..127 (32-row tiles)
  int nq = gid % NQ;

  const __bf16* a0p = xb + (size_t)(mt * 32) * DMODEL;
  const __bf16* a1p = a0p + (size_t)16 * DMODEL;
  const __bf16* wp  = wb + (size_t)(nq * 64) * DMODEL;

  v8f acc[2][4] = {};
  v16bf a0 = load_a16x32(a0p, DMODEL, lane);
  v16bf a1 = load_a16x32(a1p, DMODEL, lane);
  v16bf b0 = load_bT32x16(wp + 0 * 16 * DMODEL, DMODEL, lane);
  v16bf b1 = load_bT32x16(wp + 1 * 16 * DMODEL, DMODEL, lane);
  v16bf b2 = load_bT32x16(wp + 2 * 16 * DMODEL, DMODEL, lane);
  v16bf b3 = load_bT32x16(wp + 3 * 16 * DMODEL, DMODEL, lane);

  for (int k0 = 0; k0 < DMODEL - 32; k0 += 32) {
    int kn = k0 + 32;
    v16bf na0 = load_a16x32(a0p + kn, DMODEL, lane);
    v16bf na1 = load_a16x32(a1p + kn, DMODEL, lane);
    v16bf nb0 = load_bT32x16(wp + 0 * 16 * DMODEL + kn, DMODEL, lane);
    v16bf nb1 = load_bT32x16(wp + 1 * 16 * DMODEL + kn, DMODEL, lane);
    v16bf nb2 = load_bT32x16(wp + 2 * 16 * DMODEL + kn, DMODEL, lane);
    v16bf nb3 = load_bT32x16(wp + 3 * 16 * DMODEL + kn, DMODEL, lane);
    acc[0][0] = WMMA_BF16(a0, b0, acc[0][0]);
    acc[0][1] = WMMA_BF16(a0, b1, acc[0][1]);
    acc[0][2] = WMMA_BF16(a0, b2, acc[0][2]);
    acc[0][3] = WMMA_BF16(a0, b3, acc[0][3]);
    acc[1][0] = WMMA_BF16(a1, b0, acc[1][0]);
    acc[1][1] = WMMA_BF16(a1, b1, acc[1][1]);
    acc[1][2] = WMMA_BF16(a1, b2, acc[1][2]);
    acc[1][3] = WMMA_BF16(a1, b3, acc[1][3]);
    a0 = na0; a1 = na1; b0 = nb0; b1 = nb1; b2 = nb2; b3 = nb3;
  }
  acc[0][0] = WMMA_BF16(a0, b0, acc[0][0]);
  acc[0][1] = WMMA_BF16(a0, b1, acc[0][1]);
  acc[0][2] = WMMA_BF16(a0, b2, acc[0][2]);
  acc[0][3] = WMMA_BF16(a0, b3, acc[0][3]);
  acc[1][0] = WMMA_BF16(a1, b0, acc[1][0]);
  acc[1][1] = WMMA_BF16(a1, b1, acc[1][1]);
  acc[1][2] = WMMA_BF16(a1, b2, acc[1][2]);
  acc[1][3] = WMMA_BF16(a1, b3, acc[1][3]);

  int h = lane >> 4, n = lane & 15;
#pragma unroll
  for (int t = 0; t < 4; ++t) {
    int c = nq * 64 + t * 16 + n;
    float bv = bias[c];
    int which = c / DMODEL;
    int rem   = c % DMODEL;
    int head  = rem / HD;
    int d     = rem % HD;
#pragma unroll
    for (int s = 0; s < 2; ++s) {
#pragma unroll
      for (int e = 0; e < 8; ++e) {
        int g  = mt * 32 + s * 16 + e + h * 8;
        int bb = g >> 10;
        int ns = g & (NSEQ - 1);
        __bf16 val = (__bf16)(acc[s][t][e] + bv);
        size_t bh = (size_t)(bb * HEADS + head);
        if (which == 0)      qb[(bh * NSEQ + ns) * HDP + d] = val;
        else if (which == 1) kb[(bh * NSEQ + ns) * HDP + d] = val;
        else                 vT[(bh * HDP + d) * NSEQ + ns] = val;
      }
    }
  }
}

// ---------------------------------------------------------------------------
// Flash attention. 4 waves/block, all four waves share one (b,h): K/V tiles
// are staged into LDS once per block via async global->LDS copies, double
// buffered (issue j+32, wait current, barrier, compute, barrier).
// Q,K: (BH, NSEQ, HDP) bf16 row-major (pad cols zero). VT: (BH, HDP, NSEQ) bf16.
// OUT: (B, NSEQ, DMODEL) fp32, column head*HD + d.
__global__ __launch_bounds__(128) void flash_attn(const __bf16* __restrict__ Q,
                                                  const __bf16* __restrict__ K,
                                                  const __bf16* __restrict__ VT,
                                                  float* __restrict__ OUT,
                                                  float scale) {
  __shared__ __bf16 ldsK[2][32 * HDP];   // 2 x 6KB : rows j0..j0+31 of K
  __shared__ __bf16 ldsV[2][HDP * 32];   // 2 x 6KB : VT[d][j0..j0+31]
  __shared__ __bf16 ldsP[4][16 * 32];    // per-wave P transpose bounce

  const int tid  = threadIdx.x;
  const int w    = tid >> 5;
  const int lane = tid & 31;
  const int gid  = blockIdx.x * 4 + w;
  const int bh   = gid >> 6;             // 64 row-tiles per (b,h)
  const int mt   = gid & 63;
  const int head = bh & (HEADS - 1);
  const int bb   = bh >> 4;
  const int h = lane >> 4, n = lane & 15;
  __bf16* ldsp = ldsP[w];

  const __bf16* Qb = Q  + (size_t)bh * NSEQ * HDP + (size_t)(mt * 16) * HDP;
  const __bf16* Kb = K  + (size_t)bh * NSEQ * HDP;
  const __bf16* Vb = VT + (size_t)bh * HDP * NSEQ;

  const unsigned kbase0 = (unsigned)(size_t)(const void*)&ldsK[0][0];
  const unsigned kbase1 = (unsigned)(size_t)(const void*)&ldsK[1][0];
  const unsigned vbase0 = (unsigned)(size_t)(const void*)&ldsV[0][0];
  const unsigned vbase1 = (unsigned)(size_t)(const void*)&ldsV[1][0];

  // stage K (contiguous 6KB) + V (96 rows x 64B) for block column j0
  auto stage = [&](int buf, int j0) {
    unsigned kb_ = buf ? kbase1 : kbase0;
    unsigned vb_ = buf ? vbase1 : vbase0;
    const __bf16* ksrc = Kb + (size_t)j0 * HDP;
#pragma unroll
    for (int i = 0; i < 3; ++i) {
      int c = tid + (i << 7);                       // 0..383
      async_copy16(kb_ + (unsigned)(c * 16), ksrc + (size_t)c * 8);
      int d = c >> 2, p = c & 3;
      async_copy16(vb_ + (unsigned)(d * 64 + p * 16),
                   Vb + (size_t)d * NSEQ + j0 + p * 8);
    }
  };

  v16bf qf[3];
#pragma unroll
  for (int c = 0; c < 3; ++c) qf[c] = load_a16x32(Qb + c * 32, HDP, lane);

  float mrow[8], lrow[8], corr[8];
#pragma unroll
  for (int e = 0; e < 8; ++e) { mrow[e] = -1e30f; lrow[e] = 0.f; }
  v8f acc[5] = {};

  stage(0, 0);
  for (int j0 = 0; j0 < NSEQ; j0 += 32) {
    int cur = (j0 >> 5) & 1;
    if (j0 + 32 < NSEQ) { stage(cur ^ 1, j0 + 32); S_WAIT_ASYNC_6(); }
    else                { S_WAIT_ASYNC_0(); }
    __syncthreads();

    const __bf16* kbuf = ldsK[cur];
    const __bf16* vbuf = ldsV[cur];

    // S = q * k^T for 32 key columns (two 16x16 tiles), K frags from LDS
    v8f s0 = {}, s1 = {};
#pragma unroll
    for (int c = 0; c < 3; ++c) {
      v16bf bk0 = *(const v16bf*)(kbuf + (size_t)n * HDP + c * 32 + h * 16);
      v16bf bk1 = *(const v16bf*)(kbuf + (size_t)(16 + n) * HDP + c * 32 + h * 16);
      s0 = WMMA_BF16(qf[c], bk0, s0);
      s1 = WMMA_BF16(qf[c], bk1, s1);
    }
    // online softmax across the 32 new columns
#pragma unroll
    for (int e = 0; e < 8; ++e) {
      float a0 = s0[e] * scale, a1 = s1[e] * scale;
      float mx = fmaxf(a0, a1);
#pragma unroll
      for (int msk = 1; msk < 16; msk <<= 1) mx = fmaxf(mx, __shfl_xor(mx, msk, 32));
      float mnew = fmaxf(mrow[e], mx);
      float p0 = __expf(a0 - mnew);
      float p1 = __expf(a1 - mnew);
      float cr = __expf(mrow[e] - mnew);
      float rs = p0 + p1;
#pragma unroll
      for (int msk = 1; msk < 16; msk <<= 1) rs += __shfl_xor(rs, msk, 32);
      lrow[e] = lrow[e] * cr + rs;
      mrow[e] = mnew;
      corr[e] = cr;
      int m = e + h * 8;
      ldsp[m * 32 + n]      = (__bf16)p0;
      ldsp[m * 32 + 16 + n] = (__bf16)p1;
    }
    asm volatile("s_wait_dscnt 0" ::: "memory");
    // read P back as an A-fragment (16x32)
    v16bf pa;
    {
      int r = lane & 15;
      int base = r * 32 + h * 16;
#pragma unroll
      for (int i = 0; i < 8; ++i) { pa[i] = ldsp[base + i]; pa[i + 8] = ldsp[base + 8 + i]; }
    }
    // acc = acc*corr + P @ V ; 5 N-tiles cover hd=72 (<=80), V frags from LDS
#pragma unroll
    for (int t = 0; t < 5; ++t) {
#pragma unroll
      for (int e = 0; e < 8; ++e) acc[t][e] *= corr[e];
      v16bf bv = *(const v16bf*)(vbuf + (size_t)(t * 16 + n) * 32 + h * 16);
      acc[t] = WMMA_BF16(pa, bv, acc[t]);
    }
    __syncthreads();   // before next iteration overwrites `cur` buffer
  }

  float inv[8];
#pragma unroll
  for (int e = 0; e < 8; ++e) inv[e] = 1.0f / lrow[e];
#pragma unroll
  for (int t = 0; t < 5; ++t) {
    int d = t * 16 + n;
    if (d < HD) {
#pragma unroll
      for (int e = 0; e < 8; ++e) {
        int ns = mt * 16 + e + h * 8;
        OUT[((size_t)bb * NSEQ + ns) * DMODEL + head * HD + d] = acc[t][e] * inv[e];
      }
    }
  }
}

// ---------------------------------------------------------------------------
// Proj GEMM: out[g, c] = xo[g, :] . proj_w[c, :] + proj_b[c]  (fp32 out)
// Same pipelined 32x64 wave tile as qkv_gemm.
__global__ __launch_bounds__(256) void proj_gemm(const __bf16* __restrict__ xo,
                                                 const __bf16* __restrict__ pw,
                                                 const float*  __restrict__ bias,
                                                 float* __restrict__ out) {
  const int NQ = DMODEL / 64;                      // 18
  int gid  = blockIdx.x * 8 + (threadIdx.x >> 5);
  int lane = threadIdx.x & 31;
  int mt = gid / NQ;
  int nq = gid % NQ;

  const __bf16* a0p = xo + (size_t)(mt * 32) * DMODEL;
  const __bf16* a1p = a0p + (size_t)16 * DMODEL;
  const __bf16* wp  = pw + (size_t)(nq * 64) * DMODEL;

  v8f acc[2][4] = {};
  v16bf a0 = load_a16x32(a0p, DMODEL, lane);
  v16bf a1 = load_a16x32(a1p, DMODEL, lane);
  v16bf b0 = load_bT32x16(wp + 0 * 16 * DMODEL, DMODEL, lane);
  v16bf b1 = load_bT32x16(wp + 1 * 16 * DMODEL, DMODEL, lane);
  v16bf b2 = load_bT32x16(wp + 2 * 16 * DMODEL, DMODEL, lane);
  v16bf b3 = load_bT32x16(wp + 3 * 16 * DMODEL, DMODEL, lane);

  for (int k0 = 0; k0 < DMODEL - 32; k0 += 32) {
    int kn = k0 + 32;
    v16bf na0 = load_a16x32(a0p + kn, DMODEL, lane);
    v16bf na1 = load_a16x32(a1p + kn, DMODEL, lane);
    v16bf nb0 = load_bT32x16(wp + 0 * 16 * DMODEL + kn, DMODEL, lane);
    v16bf nb1 = load_bT32x16(wp + 1 * 16 * DMODEL + kn, DMODEL, lane);
    v16bf nb2 = load_bT32x16(wp + 2 * 16 * DMODEL + kn, DMODEL, lane);
    v16bf nb3 = load_bT32x16(wp + 3 * 16 * DMODEL + kn, DMODEL, lane);
    acc[0][0] = WMMA_BF16(a0, b0, acc[0][0]);
    acc[0][1] = WMMA_BF16(a0, b1, acc[0][1]);
    acc[0][2] = WMMA_BF16(a0, b2, acc[0][2]);
    acc[0][3] = WMMA_BF16(a0, b3, acc[0][3]);
    acc[1][0] = WMMA_BF16(a1, b0, acc[1][0]);
    acc[1][1] = WMMA_BF16(a1, b1, acc[1][1]);
    acc[1][2] = WMMA_BF16(a1, b2, acc[1][2]);
    acc[1][3] = WMMA_BF16(a1, b3, acc[1][3]);
    a0 = na0; a1 = na1; b0 = nb0; b1 = nb1; b2 = nb2; b3 = nb3;
  }
  acc[0][0] = WMMA_BF16(a0, b0, acc[0][0]);
  acc[0][1] = WMMA_BF16(a0, b1, acc[0][1]);
  acc[0][2] = WMMA_BF16(a0, b2, acc[0][2]);
  acc[0][3] = WMMA_BF16(a0, b3, acc[0][3]);
  acc[1][0] = WMMA_BF16(a1, b0, acc[1][0]);
  acc[1][1] = WMMA_BF16(a1, b1, acc[1][1]);
  acc[1][2] = WMMA_BF16(a1, b2, acc[1][2]);
  acc[1][3] = WMMA_BF16(a1, b3, acc[1][3]);

  int h = lane >> 4, n = lane & 15;
#pragma unroll
  for (int t = 0; t < 4; ++t) {
    int c = nq * 64 + t * 16 + n;
    float bv = bias[c];
#pragma unroll
    for (int s = 0; s < 2; ++s) {
#pragma unroll
      for (int e = 0; e < 8; ++e) {
        int g = mt * 32 + s * 16 + e + h * 8;
        out[(size_t)g * DMODEL + c] = acc[s][t][e] + bv;
      }
    }
  }
}

// ---------------------------------------------------------------------------
// distill loss = mean((xout - ostar)^2)
__global__ __launch_bounds__(256) void loss_partial(const float* __restrict__ a,
                                                    const float* __restrict__ b,
                                                    float* __restrict__ part, int n) {
  __shared__ float sm[256];
  float s = 0.f;
  for (int i = blockIdx.x * 256 + threadIdx.x; i < n; i += gridDim.x * 256) {
    float d = a[i] - b[i];
    s += d * d;
  }
  sm[threadIdx.x] = s;
  __syncthreads();
  for (int st = 128; st > 0; st >>= 1) {
    if (threadIdx.x < st) sm[threadIdx.x] += sm[threadIdx.x + st];
    __syncthreads();
  }
  if (threadIdx.x == 0) part[blockIdx.x] = sm[0];
}

__global__ __launch_bounds__(256) void loss_final(const float* __restrict__ part,
                                                  float* __restrict__ out, int nb, float invn) {
  __shared__ float sm[256];
  float s = 0.f;
  for (int i = threadIdx.x; i < nb; i += 256) s += part[i];
  sm[threadIdx.x] = s;
  __syncthreads();
  for (int st = 128; st > 0; st >>= 1) {
    if (threadIdx.x < st) sm[threadIdx.x] += sm[threadIdx.x + st];
    __syncthreads();
  }
  if (threadIdx.x == 0) out[0] = sm[0] * invn;
}

// ---------------------------------------------------------------------------
extern "C" void kernel_launch(void* const* d_in, const int* in_sizes, int n_in,
                              void* d_out, int out_size, void* d_ws, size_t ws_size,
                              hipStream_t stream) {
  const float* x      = (const float*)d_in[0];
  const float* enc_k  = (const float*)d_in[1];
  const float* enc_v  = (const float*)d_in[2];
  const float* qkv_w  = (const float*)d_in[3];
  const float* qkv_b  = (const float*)d_in[4];
  const float* proj_w = (const float*)d_in[5];
  const float* proj_b = (const float*)d_in[6];
  // d_in[7] = stage (always 2 per setup_inputs)

  char* ws = (char*)d_ws;
  size_t off = 0;
  auto take = [&](size_t bytes) { char* p = ws + off; off += bytes; return p; };

  const size_t nX   = (size_t)BATCH * NSEQ * DMODEL;        // 4,718,592
  const size_t nW   = (size_t)3 * DMODEL * DMODEL;          // 3,981,312
  const size_t nP   = (size_t)DMODEL * DMODEL;              // 1,327,104
  const size_t nQKV = (size_t)BH * NSEQ * HDP;              // 6,291,456 (per buffer)

  __bf16* xb   = (__bf16*)take(nX * 2);
  __bf16* wb   = (__bf16*)take(nW * 2);
  __bf16* pb   = (__bf16*)take(nP * 2);
  __bf16* qb   = (__bf16*)take(nQKV * 2);
  __bf16* kb   = (__bf16*)take(nQKV * 2);
  __bf16* vT   = (__bf16*)take(nQKV * 2);
  __bf16* ekb  = (__bf16*)take(nQKV * 2);
  __bf16* evT  = (__bf16*)take(nQKV * 2);
  float*  xout = (float*)take(nX * 4);
  float*  ostar= (float*)take(nX * 4);
  __bf16* xob  = (__bf16*)take(nX * 2);
  float*  part = (float*)take(1024 * 4);

  // zero the padded bf16 q/k/v/enc buffers (pads must be zero)
  hipMemsetAsync(qb, 0, 5 * nQKV * 2, stream);

  cvt_f32_bf16<<<(int)((nX + 255) / 256), 256, 0, stream>>>(x, xb, (int)nX);
  cvt_f32_bf16<<<(int)((nW + 255) / 256), 256, 0, stream>>>(qkv_w, wb, (int)nW);
  cvt_f32_bf16<<<(int)((nP + 255) / 256), 256, 0, stream>>>(proj_w, pb, (int)nP);
  enc_convert<<<(int)((nX + 255) / 256), 256, 0, stream>>>(enc_k, enc_v, ekb, evT);

  // QKV GEMM: 128 m-tiles(32 rows) * 54 n-quads = 6912 waves / 8 per block
  qkv_gemm<<<6912 / 8, 256, 0, stream>>>(xb, wb, qkv_b, qb, kb, vT);

  const float scale = 0.11785113019775793f;  // 72^-0.5

  // own attention and encoder attention (BH * 64 row-tiles, 4 waves/block)
  flash_attn<<<(BH * 64) / 4, 128, 0, stream>>>(qb, kb, vT, xout, scale);
  flash_attn<<<(BH * 64) / 4, 128, 0, stream>>>(qb, ekb, evT, ostar, scale);

  // distill loss -> d_out[nX]
  loss_partial<<<1024, 256, 0, stream>>>(xout, ostar, part, (int)nX);
  loss_final<<<1, 256, 0, stream>>>(part, (float*)d_out + nX, 1024, 1.0f / (float)nX);

  // out = x_out @ proj_w^T + proj_b -> d_out[0 .. nX)
  cvt_f32_bf16<<<(int)((nX + 255) / 256), 256, 0, stream>>>(xout, xob, (int)nX);
  proj_gemm<<<2304 / 8, 256, 0, stream>>>(xob, pb, proj_b, (float*)d_out);
}
